// _LSTM_27573690040400
// MI455X (gfx1250) — matrix operand order repaired and checked by reading, not compiled, not run on hardware
//
#include <hip/hip_runtime.h>
#include <hip/hip_bf16.h>

typedef __attribute__((ext_vector_type(16))) __bf16 v16bf;
typedef __attribute__((ext_vector_type(8)))  float  v8f;

#define SEQL   200
#define BATCH  1024
#define IN_CH  256
#define HID1   512
#define HID2   384
#define K1     (IN_CH + HID1)   // 768
#define K2     (HID1 + HID2)    // 896
#define KT1    (K1 / 32)        // 24 K-tiles, layer 1
#define KT2    (K2 / 32)        // 28 K-tiles, layer 2
#define NT1    (4 * HID1 / 16)  // 128 N-tiles, layer 1
#define NT2    (4 * HID2 / 16)  // 96  N-tiles, layer 2

// workspace layout (bytes)
#define WS_W1  0
#define WS_W2  (WS_W1 + NT1 * KT1 * 1024)   // 3,145,728
#define WS_B1  (WS_W2 + NT2 * KT2 * 1024)   // +2,752,512
#define WS_B2  (WS_B1 + 4 * HID1 * 4)

__device__ __forceinline__ unsigned short f2bf(float f) {
  union { float f; unsigned u; } v; v.f = f;
  unsigned u = v.u;
  return (unsigned short)((u + 0x7FFFu + ((u >> 16) & 1u)) >> 16); // RNE
}

__device__ __forceinline__ float sigm(float x) {
  return 1.0f / (1.0f + __expf(-x));
}

// Branch-free fast tanh: tanh(x) = sign(x) * (1 - e) / (1 + e), e = exp(-2|x|).
__device__ __forceinline__ float ftanh(float x) {
  float e = __expf(-2.0f * fabsf(x));
  float r = (1.0f - e) / (1.0f + e);
  return copysignf(r, x);
}

// Byte offset of element (M, kk) of K-tile kt inside an LDS A-fragment buffer.
// A-matrix 16-bit layout (ISA 7.12.2): lanes 0-15: K 0-7 in V0-3, 16-23 in V4-7;
// lanes 16-31: K 8-15 in V0-3, 24-31 in V4-7.  Fragment = 1024B, lane stride 32B.
__device__ __forceinline__ int a_off(int kt, int M, int kk) {
  int lane = M + (((kk & 15) >= 8) ? 16 : 0);
  int v = ((kk >= 16) ? 4 : 0) + ((kk & 7) >> 1);
  return kt * 1024 + lane * 32 + v * 4 + (kk & 1) * 2;
}

template <int NJT>
__device__ __forceinline__ void zacc(v8f (&acc)[NJT][4]) {
#pragma unroll
  for (int q = 0; q < NJT; ++q)
#pragma unroll
    for (int g = 0; g < 4; ++g) {
      v8f z = {0.f, 0.f, 0.f, 0.f, 0.f, 0.f, 0.f, 0.f};
      acc[q][g] = z;
    }
}

// One GEMM pass: NJT adjacent 16-wide N tiles for each of the 4 gates.
// ntile(g, q) = g*ntN + ntile0 + q.  B fragments batch-loaded with 32-bit
// byte offsets off a uniform base (SGPR base + VGPR offset form), then consumed.
template <int NJT, int KTN>
__device__ __forceinline__ void gemm_pass(v8f (&acc)[NJT][4],
                                          const unsigned char* __restrict__ wbase,
                                          const unsigned char* Abuf,
                                          int ntile0, int ntN, int lane) {
  const unsigned lane32 = (unsigned)lane * 32u;
#pragma unroll 1
  for (int kt = 0; kt < KTN; ++kt) {
    v16bf a = *((const v16bf*)(Abuf + kt * 1024) + lane);
    v16bf bb[NJT * 4];
#pragma unroll
    for (int g = 0; g < 4; ++g)
#pragma unroll
      for (int q = 0; q < NJT; ++q) {
        unsigned off = (unsigned)((g * ntN + ntile0 + q) * KTN + kt) * 1024u + lane32;
        bb[g * NJT + q] = *(const v16bf*)(wbase + off);
      }
#pragma unroll
    for (int g = 0; g < 4; ++g)
#pragma unroll
      for (int q = 0; q < NJT; ++q)
        acc[q][g] = __builtin_amdgcn_wmma_f32_16x16x32_bf16(
            false, a, false, bb[g * NJT + q], (short)0, acc[q][g], false, false);
  }
}

// LSTM pointwise update for NJT j-tiles; c kept in registers, h stashed as bf16
// bits in registers (scattered to LDS later, after the read barrier).
template <int NJT>
__device__ __forceinline__ void cell_update(v8f (&acc)[NJT][4],
                                            float (*creg)[8], unsigned short (*hs)[8],
                                            const float* __restrict__ bias, int H,
                                            int j0, int ln) {
#pragma unroll
  for (int q = 0; q < NJT; ++q) {
    int j = j0 + q * 16 + ln;
    float bI = bias[j], bF = bias[H + j], bG = bias[2 * H + j], bO = bias[3 * H + j];
#pragma unroll
    for (int r = 0; r < 8; ++r) {
      float gi = acc[q][0][r] + bI;
      float gf = acc[q][1][r] + bF;
      float gg = acc[q][2][r] + bG;
      float go = acc[q][3][r] + bO;
      float c = sigm(gf) * creg[q][r] + sigm(gi) * ftanh(gg);
      creg[q][r] = c;
      hs[q][r] = f2bf(sigm(go) * ftanh(c));
    }
  }
}

// Pack fp32 [W_ih | W_hh] rows into bf16 WMMA B-fragments:
// B 32x16 layout: lane<16 -> col=lane, K 0-7 then 16-23; lane>=16 -> col=lane-16, K 8-15 then 24-31.
__global__ void pack_w(const float* __restrict__ Wih, const float* __restrict__ Whh,
                       unsigned short* __restrict__ dst,
                       int kin, int khid, int kts, int total) {
  int gid = blockIdx.x * 256 + threadIdx.x;
  if (gid >= total) return;
  int frag = gid >> 9, idx = gid & 511;
  int ntile = frag / kts, kt = frag - ntile * kts;
  int lane = idx >> 4, rem = idx & 15;
  int K = kt * 32 + ((lane & 16) ? 8 : 0) + rem + ((rem & 8) ? 8 : 0);
  int n = ntile * 16 + (lane & 15);
  float s = (K < kin) ? Wih[n * kin + K] : Whh[n * khid + (K - kin)];
  dst[gid] = f2bf(s);
}

__global__ void bias_k(const float* bi1, const float* bh1,
                       const float* bi2, const float* bh2,
                       float* b1, float* b2) {
  int i = blockIdx.x * 256 + threadIdx.x;
  if (i < 4 * HID1) b1[i] = bi1[i] + bh1[i];
  if (i < 4 * HID2) b2[i] = bi2[i] + bh2[i];
}

__global__ __launch_bounds__(256, 1)
void lstm_persistent(const float* __restrict__ x,
                     const unsigned char* __restrict__ w1p,
                     const unsigned char* __restrict__ w2p,
                     const float* __restrict__ b1,
                     const float* __restrict__ b2,
                     const float* __restrict__ Wl1,
                     const float* __restrict__ bl1,
                     const float* __restrict__ Wl2,
                     const float* __restrict__ bl2,
                     float* __restrict__ out) {
  __shared__ __align__(32) unsigned char smem[(KT1 + KT2) * 1024]; // 52 KB
  unsigned char* A1 = smem;                 // [x_t(256) | h1(512)] bf16 fragments
  unsigned char* A2 = smem + KT1 * 1024;    // [h1(512)  | h2(384)] bf16 fragments
  float* hF = (float*)smem;                 // reused at end: 16*384 f32 == 24 KB

  const int tid = threadIdx.x;
  const int wg = blockIdx.x;
  const int wv = tid >> 5;        // wave 0..7
  const int lane = tid & 31;
  const int ln = lane & 15;       // column within 16-wide N tile
  const int mhalf = lane >> 4;    // 0: rows 0-7, 1: rows 8-15

  // zero LDS (h1/h2 regions must start at 0)
  for (int i = tid; i < (KT1 + KT2) * 64; i += 256)
    ((float4*)smem)[i] = make_float4(0.f, 0.f, 0.f, 0.f);

  float c1r[4][8], c2r[3][8];
  unsigned short h1s[4][8], h2s[3][8];
#pragma unroll
  for (int jt = 0; jt < 4; ++jt)
#pragma unroll
    for (int r = 0; r < 8; ++r) c1r[jt][r] = 0.f;
#pragma unroll
  for (int jt = 0; jt < 3; ++jt)
#pragma unroll
    for (int r = 0; r < 8; ++r) c2r[jt][r] = 0.f;

  const int jb1 = wv * 64;  // layer-1 hidden slice base for this wave
  const int jb2 = wv * 48;  // layer-2 hidden slice base

#pragma unroll 1
  for (int t = 0; t < SEQL; ++t) {
    // ---- stage 1: stage x_t tile into A1 (bf16, A-fragment layout) ----
    {
      int m = tid >> 4;
      int k0 = (tid & 15) << 4;
      const float4* xp4 =
          (const float4*)(x + (((size_t)t * BATCH) + (size_t)wg * 16 + m) * IN_CH + k0);
#pragma unroll
      for (int u4 = 0; u4 < 4; ++u4) {
        float4 v = xp4[u4];
        float vv[4] = {v.x, v.y, v.z, v.w};
#pragma unroll
        for (int e = 0; e < 4; ++e) {
          int k = k0 + u4 * 4 + e;
          *(unsigned short*)(A1 + a_off(k >> 5, m, k & 31)) = f2bf(vv[e]);
        }
      }
    }
    __syncthreads();

    // ---- stage 2: layer-1 GEMM in two passes of 2 j-tiles each ----
    {
      v8f acc[2][4];
      zacc<2>(acc);
      gemm_pass<2, KT1>(acc, w1p, A1, wv * 4 + 0, 32, lane);
      cell_update<2>(acc, c1r + 0, h1s + 0, b1, HID1, jb1 + 0, ln);
      zacc<2>(acc);
      gemm_pass<2, KT1>(acc, w1p, A1, wv * 4 + 2, 32, lane);
      cell_update<2>(acc, c1r + 2, h1s + 2, b1, HID1, jb1 + 32, ln);
    }
    __syncthreads();  // all waves done reading A1

    // ---- stage 3: scatter stashed h1 -> A1(K=256+j) & A2(K=j) ----
#pragma unroll
    for (int jt = 0; jt < 4; ++jt) {
      int j = jb1 + jt * 16 + ln;
#pragma unroll
      for (int r = 0; r < 8; ++r) {
        int M = r + 8 * mhalf;
        unsigned short hb = h1s[jt][r];
        int Ka = IN_CH + j;
        *(unsigned short*)(A1 + a_off(Ka >> 5, M, Ka & 31)) = hb;
        *(unsigned short*)(A2 + a_off(j >> 5, M, j & 31)) = hb;
      }
    }
    __syncthreads();  // A2 h1 region ready

    // ---- stage 4: layer-2 GEMM in passes of 2 + 1 j-tiles ----
    {
      v8f acc[2][4];
      zacc<2>(acc);
      gemm_pass<2, KT2>(acc, w2p, A2, wv * 3 + 0, 24, lane);
      cell_update<2>(acc, c2r + 0, h2s + 0, b2, HID2, jb2 + 0, ln);
      v8f acc1[1][4];
      zacc<1>(acc1);
      gemm_pass<1, KT2>(acc1, w2p, A2, wv * 3 + 2, 24, lane);
      cell_update<1>(acc1, c2r + 2, h2s + 2, b2, HID2, jb2 + 32, ln);
    }
    __syncthreads();  // all waves done reading A2 (h2_prev)

    // ---- stage 5: scatter stashed h2 -> A2(K=512+j); fp32 copy on last step ----
#pragma unroll
    for (int jt = 0; jt < 3; ++jt) {
      int j = jb2 + jt * 16 + ln;
#pragma unroll
      for (int r = 0; r < 8; ++r) {
        int M = r + 8 * mhalf;
        unsigned short hb = h2s[jt][r];
        int Ka = HID1 + j;
        *(unsigned short*)(A2 + a_off(Ka >> 5, M, Ka & 31)) = hb;
        if (t == SEQL - 1) {
          union { unsigned u; float f; } cv;
          cv.u = ((unsigned)hb) << 16;
          hF[M * HID2 + j] = cv.f;
        }
      }
    }
    __syncthreads();
  }

  // ---- head: y = (h2 @ Wl1^T + bl1) @ Wl2^T + bl2 ----
  if (tid < 16) {
    int m = tid;
    float y = bl2[0];
    for (int q = 0; q < 64; ++q) {
      float s = bl1[q];
      const float* wr = Wl1 + q * HID2;
      const float* hr = hF + m * HID2;
      for (int jj = 0; jj < HID2; ++jj) s += hr[jj] * wr[jj];
      y += Wl2[q] * s;
    }
    out[wg * 16 + m] = y;
  }
}

extern "C" void kernel_launch(void* const* d_in, const int* in_sizes, int n_in,
                              void* d_out, int out_size, void* d_ws, size_t ws_size,
                              hipStream_t stream) {
  const float* x      = (const float*)d_in[0];
  const float* W_ih1  = (const float*)d_in[1];
  const float* W_hh1  = (const float*)d_in[2];
  const float* b_ih1  = (const float*)d_in[3];
  const float* b_hh1  = (const float*)d_in[4];
  const float* W_ih2  = (const float*)d_in[5];
  const float* W_hh2  = (const float*)d_in[6];
  const float* b_ih2  = (const float*)d_in[7];
  const float* b_hh2  = (const float*)d_in[8];
  const float* W_lin1 = (const float*)d_in[9];
  const float* b_lin1 = (const float*)d_in[10];
  const float* W_lin2 = (const float*)d_in[11];
  const float* b_lin2 = (const float*)d_in[12];

  unsigned char* ws = (unsigned char*)d_ws;
  unsigned short* w1p = (unsigned short*)(ws + WS_W1);
  unsigned short* w2p = (unsigned short*)(ws + WS_W2);
  float* b1 = (float*)(ws + WS_B1);
  float* b2 = (float*)(ws + WS_B2);

  int tot1 = NT1 * KT1 * 512;
  pack_w<<<(tot1 + 255) / 256, 256, 0, stream>>>(W_ih1, W_hh1, w1p, IN_CH, HID1, KT1, tot1);
  int tot2 = NT2 * KT2 * 512;
  pack_w<<<(tot2 + 255) / 256, 256, 0, stream>>>(W_ih2, W_hh2, w2p, HID1, HID2, KT2, tot2);
  bias_k<<<8, 256, 0, stream>>>(b_ih1, b_hh1, b_ih2, b_hh2, b1, b2);

  lstm_persistent<<<BATCH / 16, 256, 0, stream>>>(
      x, (const unsigned char*)w1p, (const unsigned char*)w2p, b1, b2,
      W_lin1, b_lin1, W_lin2, b_lin2, (float*)d_out);
}
